// Decoder_58772332478698
// MI455X (gfx1250) — compile-verified, matrix operand order checked by reading, not run on hardware
//
#include <hip/hip_runtime.h>
#include <cstddef>

// ---------------- model dims ----------------
#define B_    64
#define LEN   24      // L
#define TT    24      // T
#define DA    1024    // D_DEC = D_ENC = D_ALIGN = 2*D_OUT
#define DW    512     // D_WEMB = D_OUT

typedef __attribute__((ext_vector_type(16))) __bf16 v16bf;
typedef __attribute__((ext_vector_type(8)))  __bf16 v8bf;
typedef __attribute__((ext_vector_type(8)))  float  v8f;

enum { GF_ACC = 1, GF_BIAS = 2 };

// ============================================================
// GEMM: C[M,N](f32) (+)= A[M,K](bf16, row stride lda) @ Wt[N,K](bf16) (+bias)
// One 16x16 tile per wave, K in steps of 32 via v_wmma_f32_16x16x32_bf16.
// A-frag per CDNA5 ISA: lane<16 -> m=lane, k elems {0..7,16..23}; lane>=16 ->
// m=lane-16, k elems {8..15,24..31}.  B-frag symmetric with n=lane&15 (Wt is
// pre-transposed [N,K] so both frags are two 16-byte loads per lane).
// C/D: lane&15 = n, vgpr v -> m = v + 8*(lane>=16).
// Requires M%16==0, N%16==0, K%32==0 and grid = (M/16 * N/16)/8 full blocks.
// ============================================================
__global__ __launch_bounds__(256) void gemm_wmma_bf16(
    const __bf16* __restrict__ A, int lda,
    const __bf16* __restrict__ Wt,
    const float* __restrict__ bias,
    float* __restrict__ C,
    int M, int N, int K, int flags)
{
  const int lane   = threadIdx.x & 31;
  const int wave   = threadIdx.x >> 5;
  const int tilesN = N >> 4;
  const int gw     = blockIdx.x * 8 + wave;
  const int tm     = gw / tilesN;
  const int tn     = gw - tm * tilesN;
  const int half   = lane >> 4;     // 0|1
  const int r      = lane & 15;

  const __bf16* aP = A  + (size_t)(tm * 16 + r) * lda + half * 8;
  const __bf16* bP = Wt + (size_t)(tn * 16 + r) * K   + half * 8;

  float* cP = C + (size_t)(tm * 16 + half * 8) * N + tn * 16 + r;
  v8f acc;
  if (flags & GF_ACC) {
#pragma unroll
    for (int v = 0; v < 8; ++v) acc[v] = cP[(size_t)v * N];
  } else {
#pragma unroll
    for (int v = 0; v < 8; ++v) acc[v] = 0.f;
  }

  for (int kb = 0; kb < K; kb += 32) {
    v8bf alo = *(const v8bf*)(aP + kb);
    v8bf ahi = *(const v8bf*)(aP + kb + 16);
    v8bf blo = *(const v8bf*)(bP + kb);
    v8bf bhi = *(const v8bf*)(bP + kb + 16);
    v16bf a, b;
#pragma unroll
    for (int i = 0; i < 8; ++i) {
      a[i] = alo[i]; a[i + 8] = ahi[i];
      b[i] = blo[i]; b[i + 8] = bhi[i];
    }
    acc = __builtin_amdgcn_wmma_f32_16x16x32_bf16(
        /*neg_a=*/false, a, /*neg_b=*/false, b,
        /*c_mod=*/(short)0, acc, /*reuse_a=*/false, /*reuse_b=*/false);
  }

  if (flags & GF_BIAS) {
    float bv = bias[tn * 16 + r];
#pragma unroll
    for (int v = 0; v < 8; ++v) acc[v] += bv;
  }
#pragma unroll
  for (int v = 0; v < 8; ++v) cP[(size_t)v * N] = acc[v];
}

// ------------------------------------------------------------
// dst[n*dstStride + dstOff + k] = bf16(src[k*N + n])   (K,N multiples of 32)
// ------------------------------------------------------------
__global__ __launch_bounds__(256) void transpose_to_bf16(
    const float* __restrict__ src, __bf16* __restrict__ dst,
    int K, int N, int dstStride, int dstOff)
{
  __shared__ float tile[32][33];
  const int k0 = blockIdx.x * 32, n0 = blockIdx.y * 32;
  for (int i = threadIdx.y; i < 32; i += 8)
    tile[i][threadIdx.x] = src[(size_t)(k0 + i) * N + n0 + threadIdx.x];
  __syncthreads();
  for (int i = threadIdx.y; i < 32; i += 8)
    dst[(size_t)(n0 + i) * dstStride + dstOff + k0 + threadIdx.x] =
        (__bf16)tile[threadIdx.x][i];
}

__global__ void embed_to_bf16(const float* __restrict__ emb,
                              const int* __restrict__ ys,
                              __bf16* __restrict__ out)
{
  const int row = blockIdx.x;           // T*B rows
  const int tok = ys[row];
  out[(size_t)row * DW + threadIdx.x] = (__bf16)emb[(size_t)tok * DW + threadIdx.x];
}

__global__ void f32_to_bf16(const float* __restrict__ src,
                            __bf16* __restrict__ dst, int n)
{
  int i = blockIdx.x * 256 + threadIdx.x;
  if (i < n) dst[i] = (__bf16)src[i];
}

__global__ void copy_f32(float* __restrict__ dst, const float* __restrict__ src, int n)
{
  int i = blockIdx.x * 256 + threadIdx.x;
  if (i < n) dst[i] = src[i];
}

__global__ void add3_f32(float* __restrict__ dst, const float* __restrict__ a,
                         const float* __restrict__ b, const float* __restrict__ c, int n)
{
  int i = blockIdx.x * 256 + threadIdx.x;
  if (i < n) dst[i] = a[i] + b[i] + (c ? c[i] : 0.f);
}

// ------------------------------------------------------------
// GRU gates z,r:  hzr[b, 0:1024]=h@Uz(+...), hzr[b,1024:2048]=h@Ur(+...)
// optional xg row = precomputed x@W + bias (cols {0,1024}+e, stride xgStride)
// rsOut <- bf16(r * h)   (written into a K-concat A buffer)
// ------------------------------------------------------------
__global__ void gru_zr(const float* __restrict__ hzr,
                       const float* __restrict__ xg, int xgStride,
                       const float* __restrict__ h,
                       float* __restrict__ zOut,
                       __bf16* __restrict__ rsOut, int rsStride, int rsOff)
{
  const int b = blockIdx.y;
  const int e = blockIdx.x * 256 + threadIdx.x;
  float xz = 0.f, xr = 0.f;
  if (xg) {
    xz = xg[(size_t)b * xgStride + e];
    xr = xg[(size_t)b * xgStride + DA + e];
  }
  float z = 1.f / (1.f + expf(-(xz + hzr[(size_t)b * 2048 + e])));
  float r = 1.f / (1.f + expf(-(xr + hzr[(size_t)b * 2048 + DA + e])));
  zOut[(size_t)b * DA + e] = z;
  rsOut[(size_t)b * rsStride + rsOff + e] = (__bf16)(r * h[(size_t)b * DA + e]);
}

// hc = tanh(xh + hh); hn = (1-z)h + z*hc; out = m*hn + (1-m)*h
__global__ void gru_h(const float* __restrict__ hh,
                      const float* __restrict__ xg, int xgOff, int xgStride,
                      const float* __restrict__ z, const float* __restrict__ h,
                      const float* __restrict__ ym,
                      float* __restrict__ sOut,
                      __bf16* __restrict__ sBf, int sbStride, int sbOff)
{
  const int b = blockIdx.y;
  const int e = blockIdx.x * 256 + threadIdx.x;
  float xh = xg ? xg[(size_t)b * xgStride + xgOff + e] : 0.f;
  float hc = tanhf(xh + hh[(size_t)b * DA + e]);
  float zz = z[(size_t)b * DA + e];
  float hv = h[(size_t)b * DA + e];
  float hn = (1.f - zz) * hv + zz * hc;
  float m  = ym[b];
  float o  = m * hn + (1.f - m) * hv;
  if (sOut) sOut[(size_t)b * DA + e] = o;
  if (sBf)  sBf[(size_t)b * sbStride + sbOff + e] = (__bf16)o;
}

// ------------------------------------------------------------
// esc[gw] = exp( dot(tanh(proj[b]+U[gw]), a1w) + a1b ) * mask[gw]
// one wave per (row,b).  U (uh / uh_cyk) is a one-shot 151 MB stream per step:
// read non-temporally so it does not thrash the 192 MB L2 that holds the
// recurrent GEMM weights (~24 MB bf16, reused all 24 steps).
// ------------------------------------------------------------
__global__ void attn_scores(const float* __restrict__ U,
                            const float* __restrict__ proj,
                            const float* __restrict__ a1w,
                            const float* __restrict__ a1b,
                            const float* __restrict__ mask,
                            float* __restrict__ esc, int nRB)
{
  const int gw   = blockIdx.x * 8 + (threadIdx.x >> 5);
  const int lane = threadIdx.x & 31;
  if (gw >= nRB) return;
  const int b = gw & (B_ - 1);
  const float* u = U    + (size_t)gw * DA;
  const float* p = proj + (size_t)b  * DA;
  float s = 0.f;
  for (int e = lane; e < DA; e += 32) {
    float uv = __builtin_nontemporal_load(&u[e]);   // TH=NT stream
    s += tanhf(p[e] + uv) * a1w[e];
  }
#pragma unroll
  for (int d = 16; d; d >>= 1) s += __shfl_xor(s, d, 32);
  if (lane == 0) esc[gw] = expf(s + a1b[0]) * mask[gw];
}

// outBf[b, outOff+e] = bf16( sum_r esc[r,b]*X[r,b,e] / sum_r esc[r,b] )
// X (xs_h / xs_cyk) also read non-temporally (one-shot stream per step).
__global__ void attn_combine(const float* __restrict__ esc,
                             const float* __restrict__ X,
                             __bf16* __restrict__ outBf,
                             int R, int outStride, int outOff)
{
  const int b = blockIdx.y;
  const int e = blockIdx.x * 256 + threadIdx.x;
  float wsum = 0.f, acc = 0.f;
  for (int rI = 0; rI < R; ++rI) {
    float w = esc[rI * B_ + b];           // uniform -> scalar load
    wsum += w;
    acc  += w * __builtin_nontemporal_load(&X[((size_t)rI * B_ + b) * DA + e]);
  }
  outBf[(size_t)b * outStride + outOff + e] = (__bf16)(acc / wsum);
}

__global__ void copy_to_2bf16(const float* __restrict__ src,
                              __bf16* __restrict__ d1, int d1Stride,
                              __bf16* __restrict__ d2)
{
  const int b = blockIdx.y;
  const int e = blockIdx.x * 256 + threadIdx.x;
  float v = src[(size_t)b * DA + e];
  d1[(size_t)b * d1Stride + e] = (__bf16)v;
  d2[(size_t)b * DA + e]       = (__bf16)v;
}

// out[row,j] = max(logit[row,2j], logit[row,2j+1]) * ys_mask[row]
__global__ void maxout_mask(const float* __restrict__ logit,
                            const float* __restrict__ ym,
                            float* __restrict__ out)
{
  const int row = blockIdx.x;             // T*B
  const int j   = threadIdx.x;            // 512
  float a = logit[(size_t)row * DA + 2 * j];
  float c = logit[(size_t)row * DA + 2 * j + 1];
  out[(size_t)row * DW + j] = fmaxf(a, c) * ym[row];
}

// ============================================================
extern "C" void kernel_launch(void* const* d_in, const int* in_sizes, int n_in,
                              void* d_out, int out_size, void* d_ws, size_t ws_size,
                              hipStream_t stream) {
  (void)in_sizes; (void)n_in; (void)out_size; (void)ws_size;
  // ---- inputs, setup_inputs() dict order (params flattened in insertion order)
  const float* s_tm1  = (const float*)d_in[0];   // [64,1024]
  const float* xs_h   = (const float*)d_in[1];   // [24,64,1024]
  const int*   ys     = (const int*)  d_in[2];   // [24,64]
  const float* uh     = (const float*)d_in[3];   // [24,64,1024]
  const float* xs_cyk = (const float*)d_in[4];   // [24,24,64,1024]
  const float* uh_cyk = (const float*)d_in[5];   // [24,24,64,1024]
  const float* xs_mask= (const float*)d_in[6];   // [24,64]
  const float* ys_mask= (const float*)d_in[7];   // [24,64]
  const float* cykmask= (const float*)d_in[8];   // [24,24,64]
  const float* emb    = (const float*)d_in[9];   // [30000,512]
  const float* g1_Wz  = (const float*)d_in[10];  const float* g1_Uz = (const float*)d_in[11];
  const float* g1_bz  = (const float*)d_in[12];  const float* g1_Wr = (const float*)d_in[13];
  const float* g1_Ur  = (const float*)d_in[14];  const float* g1_br = (const float*)d_in[15];
  const float* g1_Wh  = (const float*)d_in[16];  const float* g1_Uh = (const float*)d_in[17];
  const float* g1_bh  = (const float*)d_in[18];
  const float* g2_Wz  = (const float*)d_in[19];  const float* g2_Uz = (const float*)d_in[20];
  const float* g2_bz  = (const float*)d_in[21];  const float* g2_Wr = (const float*)d_in[22];
  const float* g2_Ur  = (const float*)d_in[23];  const float* g2_br = (const float*)d_in[24];
  const float* g2_Wh  = (const float*)d_in[25];  const float* g2_Uh = (const float*)d_in[26];
  const float* g2_bh  = (const float*)d_in[27];
  const float* sa_W   = (const float*)d_in[28];  const float* sa_b  = (const float*)d_in[29];
  const float* a1_W   = (const float*)d_in[30];  const float* a1_b  = (const float*)d_in[31];
  const float* as_W   = (const float*)d_in[32];  const float* as_b  = (const float*)d_in[33];
  const float* ac_W   = (const float*)d_in[34];  const float* ac_b  = (const float*)d_in[35];
  const float* ls_W   = (const float*)d_in[36];  const float* ls_b  = (const float*)d_in[37];
  const float* ly_W   = (const float*)d_in[38];  const float* ly_b  = (const float*)d_in[39];
  const float* lc_W   = (const float*)d_in[40];  const float* lc_b  = (const float*)d_in[41];

  // ---- workspace carving (256B aligned), ~68 MB total
  char* ws = (char*)d_ws;
  size_t off = 0;
  auto alloc = [&](size_t bytes) -> void* {
    void* p = (void*)(ws + off);
    off += (bytes + 255) & ~(size_t)255;
    return p;
  };
  __bf16* wt_zr1 = (__bf16*)alloc((size_t)2048 * 1024 * 2); // [Uz1;Ur1]^T  [2048n,1024k]
  __bf16* wt_h1  = (__bf16*)alloc((size_t)1024 * 1024 * 2); // Uh1^T
  __bf16* wt_x1  = (__bf16*)alloc((size_t)3072 * 512  * 2); // [Wz1;Wr1;Wh1]^T [3072n,512k]
  __bf16* wt_sa  = (__bf16*)alloc((size_t)1024 * 1024 * 2); // sa_W^T
  __bf16* wt_att = (__bf16*)alloc((size_t)1024 * 2048 * 2); // [as_W|ac_W]^T  [1024n,2048k]
  __bf16* wt_zr2 = (__bf16*)alloc((size_t)2048 * 2048 * 2); // [[Wz2|Uz2];[Wr2|Ur2]]^T
  __bf16* wt_h2  = (__bf16*)alloc((size_t)1024 * 2048 * 2); // [Wh2|Uh2]^T
  __bf16* wt_ls  = (__bf16*)alloc((size_t)1024 * 1024 * 2);
  __bf16* wt_ly  = (__bf16*)alloc((size_t)1024 * 512  * 2);
  __bf16* wt_lc  = (__bf16*)alloc((size_t)1024 * 1024 * 2);
  float*  bx1    = (float*) alloc(3072 * 4);                // bz1|br1|bh1
  float*  bzr2   = (float*) alloc(2048 * 4);                // bz2|br2
  float*  b_att  = (float*) alloc(1024 * 4);                // as_b+ac_b
  float*  b_out  = (float*) alloc(1024 * 4);                // ls_b+ly_b+lc_b
  __bf16* ysE    = (__bf16*)alloc((size_t)TT * B_ * DW * 2);
  float*  xg1    = (float*) alloc((size_t)TT * B_ * 3072 * 4); // ys_e@[Wz|Wr|Wh]+b
  float*  s_buf  = (float*) alloc((size_t)B_ * DA * 4);
  float*  s_ab   = (float*) alloc((size_t)B_ * DA * 4);     // s_above (f32)
  __bf16* sAbf   = (__bf16*)alloc((size_t)B_ * DA * 2);
  float*  t_zr   = (float*) alloc((size_t)B_ * 2048 * 4);
  float*  z1     = (float*) alloc((size_t)B_ * DA * 4);
  __bf16* rsbf   = (__bf16*)alloc((size_t)B_ * DA * 2);
  float*  t_h    = (float*) alloc((size_t)B_ * DA * 4);
  float*  proj   = (float*) alloc((size_t)B_ * DA * 4);
  float*  escS   = (float*) alloc((size_t)LEN * B_ * 4);
  float*  escC   = (float*) alloc((size_t)LEN * LEN * B_ * 4);
  __bf16* Acat   = (__bf16*)alloc((size_t)B_ * 2048 * 2);   // [attend_seq|attend_cyk] bf16
  float*  attend = (float*) alloc((size_t)B_ * DA * 4);
  __bf16* A2     = (__bf16*)alloc((size_t)B_ * 2048 * 2);   // [attend|s_above] then [attend|r*s]
  float*  t_zr2  = (float*) alloc((size_t)B_ * 2048 * 4);
  float*  z2     = (float*) alloc((size_t)B_ * DA * 4);
  float*  t_h2   = (float*) alloc((size_t)B_ * DA * 4);
  __bf16* sAll   = (__bf16*)alloc((size_t)TT * B_ * DA * 2);
  __bf16* cAll   = (__bf16*)alloc((size_t)TT * B_ * DA * 2);
  float*  logit  = (float*) alloc((size_t)TT * B_ * DA * 4);

  auto T2B = [&](const float* src, __bf16* dst, int K, int N, int stride, int koff) {
    transpose_to_bf16<<<dim3(K / 32, N / 32), dim3(32, 8), 0, stream>>>(src, dst, K, N, stride, koff);
  };
  auto GEMM = [&](const __bf16* A, int lda, const __bf16* Wt, const float* bias,
                  float* C, int M, int N, int K, int flags) {
    gemm_wmma_bf16<<<dim3((M / 16) * (N / 16) / 8), dim3(256), 0, stream>>>(
        A, lda, Wt, bias, C, M, N, K, flags);
  };

  // ---- phase 0: weight transpose+convert to bf16 [N,K]
  T2B(g1_Uz, wt_zr1, 1024, 1024, 1024, 0);
  T2B(g1_Ur, wt_zr1 + (size_t)1024 * 1024, 1024, 1024, 1024, 0);
  T2B(g1_Uh, wt_h1, 1024, 1024, 1024, 0);
  T2B(g1_Wz, wt_x1, 512, 1024, 512, 0);
  T2B(g1_Wr, wt_x1 + (size_t)1024 * 512, 512, 1024, 512, 0);
  T2B(g1_Wh, wt_x1 + (size_t)2048 * 512, 512, 1024, 512, 0);
  T2B(sa_W, wt_sa, 1024, 1024, 1024, 0);
  T2B(as_W, wt_att, 1024, 1024, 2048, 0);
  T2B(ac_W, wt_att, 1024, 1024, 2048, 1024);
  T2B(g2_Wz, wt_zr2, 1024, 1024, 2048, 0);
  T2B(g2_Uz, wt_zr2, 1024, 1024, 2048, 1024);
  T2B(g2_Wr, wt_zr2 + (size_t)1024 * 2048, 1024, 1024, 2048, 0);
  T2B(g2_Ur, wt_zr2 + (size_t)1024 * 2048, 1024, 1024, 2048, 1024);
  T2B(g2_Wh, wt_h2, 1024, 1024, 2048, 0);
  T2B(g2_Uh, wt_h2, 1024, 1024, 2048, 1024);
  T2B(ls_W, wt_ls, 1024, 1024, 1024, 0);
  T2B(ly_W, wt_ly, 512, 1024, 512, 0);
  T2B(lc_W, wt_lc, 1024, 1024, 1024, 0);
  copy_f32<<<4, 256, 0, stream>>>(bx1,        g1_bz, 1024);
  copy_f32<<<4, 256, 0, stream>>>(bx1 + 1024, g1_br, 1024);
  copy_f32<<<4, 256, 0, stream>>>(bx1 + 2048, g1_bh, 1024);
  copy_f32<<<4, 256, 0, stream>>>(bzr2,        g2_bz, 1024);
  copy_f32<<<4, 256, 0, stream>>>(bzr2 + 1024, g2_br, 1024);
  add3_f32<<<4, 256, 0, stream>>>(b_att, as_b, ac_b, nullptr, 1024);
  add3_f32<<<4, 256, 0, stream>>>(b_out, ls_b, ly_b, lc_b, 1024);

  // ---- phase 1: embedding + all-timestep GRU1 x-side gates
  embed_to_bf16<<<TT * B_, DW, 0, stream>>>(emb, ys, ysE);
  GEMM(ysE, DW, wt_x1, bx1, xg1, TT * B_, 3072, DW, GF_BIAS);

  // ---- phase 2: 24 recurrent steps
  for (int t = 0; t < TT; ++t) {
    const float* h  = (t == 0) ? s_tm1 : s_buf;
    const float* ym = ys_mask + t * B_;
    const float* xg = xg1 + (size_t)t * B_ * 3072;
    // GRU1
    f32_to_bf16<<<256, 256, 0, stream>>>(h, sAbf, B_ * DA);
    GEMM(sAbf, DA, wt_zr1, nullptr, t_zr, B_, 2048, DA, 0);
    gru_zr<<<dim3(4, B_), 256, 0, stream>>>(t_zr, xg, 3072, h, z1, rsbf, DA, 0);
    GEMM(rsbf, DA, wt_h1, nullptr, t_h, B_, DA, DA, 0);
    gru_h<<<dim3(4, B_), 256, 0, stream>>>(t_h, xg, 2048, 3072, z1, h, ym,
                                           s_ab, A2, 2048, 1024); // s_above -> f32 + bf16 in A2[:,1024:]
    // alignment projection
    GEMM(A2 + 1024, 2048, wt_sa, sa_b, proj, B_, DA, DA, GF_BIAS);
    // attention (seq + CYK): HBM-streaming kernels (NT loads)
    attn_scores<<<(LEN * B_) / 8, 256, 0, stream>>>(uh, proj, a1_W, a1_b, xs_mask, escS, LEN * B_);
    attn_scores<<<(LEN * LEN * B_) / 8, 256, 0, stream>>>(uh_cyk, proj, a1_W, a1_b, cykmask,
                                                          escC, LEN * LEN * B_);
    attn_combine<<<dim3(4, B_), 256, 0, stream>>>(escS, xs_h,   Acat, LEN,       2048, 0);
    attn_combine<<<dim3(4, B_), 256, 0, stream>>>(escC, xs_cyk, Acat, LEN * LEN, 2048, 1024);
    GEMM(Acat, 2048, wt_att, b_att, attend, B_, DA, 2048, GF_BIAS);
    copy_to_2bf16<<<dim3(4, B_), 256, 0, stream>>>(attend, A2, 2048, cAll + (size_t)t * B_ * DA);
    // GRU2 (x|h fused along K)
    GEMM(A2, 2048, wt_zr2, bzr2, t_zr2, B_, 2048, 2048, GF_BIAS);
    gru_zr<<<dim3(4, B_), 256, 0, stream>>>(t_zr2, nullptr, 0, s_ab, z2, A2, 2048, 1024);
    GEMM(A2, 2048, wt_h2, g2_bh, t_h2, B_, DA, 2048, GF_BIAS);
    gru_h<<<dim3(4, B_), 256, 0, stream>>>(t_h2, nullptr, 0, 0, z2, s_ab, ym,
                                           s_buf, sAll + (size_t)t * B_ * DA, DA, 0);
  }

  // ---- phase 3: output layer (3 accumulated GEMMs) + maxout
  GEMM(sAll, DA, wt_ls, b_out,  logit, TT * B_, DA, DA, GF_BIAS);
  GEMM(ysE,  DW, wt_ly, nullptr, logit, TT * B_, DA, DW, GF_ACC);
  GEMM(cAll, DA, wt_lc, nullptr, logit, TT * B_, DA, DA, GF_ACC);
  maxout_mask<<<TT * B_, DW, 0, stream>>>(logit, ys_mask, (float*)d_out);
}
